// SwinMAE_35296041239130
// MI455X (gfx1250) — matrix operand order repaired and checked by reading, not compiled, not run on hardware
//
#include <hip/hip_runtime.h>
#include <hip/hip_bf16.h>

typedef __attribute__((ext_vector_type(16))) _Float16 v16h;
typedef __attribute__((ext_vector_type(4)))  _Float16 h4;
typedef __attribute__((ext_vector_type(8)))  float    v8f;

#define WMMA16(a,b,c) __builtin_amdgcn_wmma_f32_16x16x32_f16(false,(a),false,(b),(short)0,(c),false,false)

static constexpr int BATCH = 32;
static constexpr int L0    = 56 * 56;
static constexpr int PREDC = 48;

// involutive k-swizzle: fragment order per lane-half is [0..7,16..23] | [8..15,24..31]
__device__ __forceinline__ int swz(int k) { return (k & 7) | ((k & 8) << 1) | ((k & 16) >> 1); }

__device__ __forceinline__ float gelu_exact(float x) {
    return 0.5f * x * (1.f + erff(x * 0.70710678118654752f));
}

// ---------------------------------------------------------------- elementwise
__global__ void fill_kernel(float* p, float v, int n) {
    int i = blockIdx.x * 256 + threadIdx.x;
    if (i < n) p[i] = v;
}

// LayerNorm: one wave per row
__global__ void ln_kernel(const float* __restrict__ in, const float* g, const float* b,
                          float* out, int M, int C) {
    int row  = blockIdx.x * 8 + (threadIdx.x >> 5);
    int lane = threadIdx.x & 31;
    if (row >= M) return;
    const float* r = in + (size_t)row * C;
    float s = 0.f, s2 = 0.f;
    for (int c = lane; c < C; c += 32) { float v = r[c]; s += v; s2 += v * v; }
    for (int m = 16; m > 0; m >>= 1) { s += __shfl_xor(s, m); s2 += __shfl_xor(s2, m); }
    float mean = s / C;
    float var  = s2 / C - mean * mean;
    float inv  = rsqrtf(var + 1e-6f);
    float* o = out + (size_t)row * C;
    for (int c = lane; c < C; c += 32) o[c] = (r[c] - mean) * inv * g[c] + b[c];
}

// ---------------------------------------------------------------- patch embed
__global__ void patch_embed_kernel(const float* __restrict__ x, const float* __restrict__ pw,
                                   const float* __restrict__ pb, float* out) {
    int idx = blockIdx.x * 256 + threadIdx.x;
    if (idx >= BATCH * L0 * 96) return;
    int e = idx % 96, t = idx / 96;
    int w = t % 56, h = (t / 56) % 56, b = t / L0;
    float acc = pb[e];
    for (int c = 0; c < 3; ++c)
        for (int p = 0; p < 4; ++p)
            for (int q = 0; q < 4; ++q)
                acc += x[((size_t)(b * 3 + c) * 224 + h * 4 + p) * 224 + w * 4 + q] *
                       pw[((e * 3 + c) * 4 + p) * 4 + q];
    out[idx] = acc;
}

// ---------------------------------------------------------------- masking
__global__ void mask_rank_kernel(const float* __restrict__ noise, float* maskBL) {
    int b = blockIdx.x, t = threadIdx.x;
    if (t >= 196) return;
    float nv = noise[b * 196 + t];
    int rank = 0;
    for (int j = 0; j < 196; ++j) {
        float oj = noise[b * 196 + j];
        if (oj < nv || (oj == nv && j < t)) ++rank;
    }
    if (rank < 147) {
        int base = (t / 14) * 224 + (t % 14) * 4;
        for (int i = 0; i < 4; ++i)
            for (int j = 0; j < 4; ++j)
                maskBL[b * L0 + base + i * 56 + j] = 0.f;
    }
}

__global__ void blend_kernel(const float* __restrict__ xe, const float* __restrict__ mtok,
                             const float* __restrict__ maskBL, float* h) {
    int idx = blockIdx.x * 256 + threadIdx.x;
    if (idx >= BATCH * L0 * 96) return;
    int c = idx % 96, bl = idx / 96;
    float m = maskBL[bl];
    h[idx] = xe[idx] * (1.f - m) + mtok[c] * m;
}

// ---------------------------------------------------------------- generic WMMA GEMM
// C = act(A*B + bias) + resid. K%32==0, N%16==0. fp32 io, f16 WMMA, fp32 acc.
// 64x64 tile; 8 waves as 4(M) x 2(N-pairs); A staged row-major, B staged transposed.
// act: 0 = identity, 1 = exact GELU. resid may be null (same layout as C).
__global__ void gemm_kernel(const float* __restrict__ A, const float* __restrict__ Bw,
                            const float* __restrict__ bias, const float* __restrict__ resid,
                            float* __restrict__ C, int M, int N, int K, int act) {
    __shared__ _Float16 As[64 * 32];   // [row][k]
    __shared__ _Float16 Bt[64 * 32];   // [nLocal][k]
    int tid = threadIdx.x, lane = tid & 31, wid = tid >> 5;
    int m0 = blockIdx.x * 64, n0 = blockIdx.y * 64;
    int wm = wid & 3, wn = wid >> 2;
    v8f acc0 = {}, acc1 = {};
    int mrow = wm * 16 + (lane & 15), half = lane >> 4;
    int nc0  = wn * 32 + (lane & 15);
    int kg   = half * 16;
    // per-thread staging coordinates
    int ra0 = tid >> 3, ca0 = (tid & 7) * 4;      // A rows 0..31 | 32..63
    int rb0 = tid >> 4, cb0 = (tid & 15) * 4;     // B k-rows 0..15 | 16..31
    const float* Ab = A + (size_t)m0 * K;
    const float* Bb = Bw + n0;
    bool interior = (m0 + 64 <= M) && (n0 + 64 <= N);

    for (int k0 = 0; k0 < K; k0 += 32) {
        if (interior) {
            // issue all four 128-bit loads, then convert/store
            float4 av0 = *(const float4*)(Ab + (size_t)ra0 * K + k0 + ca0);
            float4 av1 = *(const float4*)(Ab + (size_t)(ra0 + 32) * K + k0 + ca0);
            float4 bv0 = *(const float4*)(Bb + (size_t)(k0 + rb0) * N + cb0);
            float4 bv1 = *(const float4*)(Bb + (size_t)(k0 + rb0 + 16) * N + cb0);
            int ap = ca0;
            *(h4*)&As[ra0 * 32 + ap] =
                h4{(_Float16)av0.x, (_Float16)av0.y, (_Float16)av0.z, (_Float16)av0.w};
            *(h4*)&As[(ra0 + 32) * 32 + ap] =
                h4{(_Float16)av1.x, (_Float16)av1.y, (_Float16)av1.z, (_Float16)av1.w};
            Bt[(cb0 + 0) * 32 + rb0] = (_Float16)bv0.x;
            Bt[(cb0 + 1) * 32 + rb0] = (_Float16)bv0.y;
            Bt[(cb0 + 2) * 32 + rb0] = (_Float16)bv0.z;
            Bt[(cb0 + 3) * 32 + rb0] = (_Float16)bv0.w;
            Bt[(cb0 + 0) * 32 + rb0 + 16] = (_Float16)bv1.x;
            Bt[(cb0 + 1) * 32 + rb0 + 16] = (_Float16)bv1.y;
            Bt[(cb0 + 2) * 32 + rb0 + 16] = (_Float16)bv1.z;
            Bt[(cb0 + 3) * 32 + rb0 + 16] = (_Float16)bv1.w;
        } else {
            for (int i = tid; i < 512; i += 256) {
                int r = i >> 3, c4 = (i & 7) * 4, gr = m0 + r;
                float4 v{0.f, 0.f, 0.f, 0.f};
                if (gr < M) v = *(const float4*)(A + (size_t)gr * K + k0 + c4);
                *(h4*)&As[r * 32 + c4] =
                    h4{(_Float16)v.x, (_Float16)v.y, (_Float16)v.z, (_Float16)v.w};
            }
            for (int i = tid; i < 512; i += 256) {
                int r = i >> 4, c4 = (i & 15) * 4, gc = n0 + c4;
                float4 v{0.f, 0.f, 0.f, 0.f};
                if (gc < N) v = *(const float4*)(Bw + (size_t)(k0 + r) * N + gc);
                Bt[(c4 + 0) * 32 + r] = (_Float16)v.x;
                Bt[(c4 + 1) * 32 + r] = (_Float16)v.y;
                Bt[(c4 + 2) * 32 + r] = (_Float16)v.z;
                Bt[(c4 + 3) * 32 + r] = (_Float16)v.w;
            }
        }
        __syncthreads();
        v16h a   = *(const v16h*)&As[mrow * 32 + half * 16];
        v16h bf0 = *(const v16h*)&Bt[nc0 * 32 + kg];
        v16h bf1 = *(const v16h*)&Bt[(nc0 + 16) * 32 + kg];
        acc0 = WMMA16(a, bf0, acc0);
        acc1 = WMMA16(a, bf1, acc1);
        __syncthreads();
    }
    int col0 = n0 + nc0;
    float bv0 = (bias != nullptr && col0 < N) ? bias[col0] : 0.f;
    float bv1 = (bias != nullptr && col0 + 16 < N) ? bias[col0 + 16] : 0.f;
#pragma unroll
    for (int v = 0; v < 8; ++v) {
        int row = m0 + wm * 16 + v + 8 * half;
        if (row < M) {
            if (col0 < N) {
                float r0 = acc0[v] + bv0;
                if (act == 1) r0 = gelu_exact(r0);
                if (resid)    r0 += resid[(size_t)row * N + col0];
                C[(size_t)row * N + col0] = r0;
            }
            if (col0 + 16 < N) {
                float r1 = acc1[v] + bv1;
                if (act == 1) r1 = gelu_exact(r1);
                if (resid)    r1 += resid[(size_t)row * N + col0 + 16];
                C[(size_t)row * N + col0 + 16] = r1;
            }
        }
    }
}

// ---------------------------------------------------------------- window (un)partition
__global__ void part_kernel(const float* __restrict__ x, float* __restrict__ xw,
                            int Hd, int C, int shift) {
    int n_tot = BATCH * Hd * Hd * C;
    int idx = blockIdx.x * 256 + threadIdx.x;
    if (idx >= n_tot) return;
    int c = idx % C, t = idx / C;
    int n = t % 49, w_ = t / 49;
    int nwx = Hd / 7, nW = nwx * nwx;
    int wim = w_ % nW, b = w_ / nW;
    int wy = wim / nwx, wx = wim % nwx;
    int a = n / 7, bb = n % 7;
    int ys = (wy * 7 + a + shift) % Hd;
    int xs = (wx * 7 + bb + shift) % Hd;
    xw[idx] = x[((size_t)(b * Hd + ys) * Hd + xs) * C + c];
}

__global__ void rev_add_kernel(const float* __restrict__ shortb, const float* __restrict__ win,
                               float* __restrict__ outb, int Hd, int C, int shift) {
    int n_tot = BATCH * Hd * Hd * C;
    int idx = blockIdx.x * 256 + threadIdx.x;
    if (idx >= n_tot) return;
    int c = idx % C, t = idx / C;
    int x = t % Hd, y = (t / Hd) % Hd, b = t / (Hd * Hd);
    int ys = (y - shift + Hd) % Hd, xs = (x - shift + Hd) % Hd;
    int nwx = Hd / 7;
    int wy = ys / 7, a = ys % 7, wx = xs / 7, bb = xs % 7;
    size_t widx = (((size_t)((b * nwx + wy) * nwx + wx)) * 49 + a * 7 + bb) * C + c;
    outb[idx] = shortb[idx] + win[widx];
}

// ---------------------------------------------------------------- fused window attention
__global__ void attn_kernel(const float* __restrict__ qkv, const float* __restrict__ relb,
                            float* __restrict__ out, int nh, int nWimg, int Hd, int shift) {
    int blk = blockIdx.x;
    int h = blk % nh, w_ = blk / nh;
    int C = nh * 32;
    __shared__ _Float16 Qs[64 * 32];   // [n][d]
    __shared__ _Float16 Ks[64 * 32];   // [n][d]
    __shared__ _Float16 Vt[32 * 64];   // [d][n]
    __shared__ float    Sf[64][64];
    __shared__ _Float16 Pf[64 * 64];   // [i][j]
    int tid = threadIdx.x, lane = tid & 31, wid = tid >> 5;
    const float scale = 0.17677669529663687f;   // 32^-0.5

    {
        int n0 = tid >> 3, d4 = (tid & 7) * 4;       // chunk0: n 0..31 (always valid)
        int n1 = n0 + 32;                            // chunk1: n 32..63 (valid if <49)
        const float* tb0 = qkv + (size_t)(w_ * 49 + n0) * 3 * C + h * 32;
        float4 q0 = *(const float4*)(tb0 + d4);
        float4 k0 = *(const float4*)(tb0 + C + d4);
        float4 v0 = *(const float4*)(tb0 + 2 * C + d4);
        float4 q1{0.f, 0.f, 0.f, 0.f}, k1{0.f, 0.f, 0.f, 0.f}, v1{0.f, 0.f, 0.f, 0.f};
        if (n1 < 49) {
            const float* tb1 = qkv + (size_t)(w_ * 49 + n1) * 3 * C + h * 32;
            q1 = *(const float4*)(tb1 + d4);
            k1 = *(const float4*)(tb1 + C + d4);
            v1 = *(const float4*)(tb1 + 2 * C + d4);
        }
        int dp = d4;
        *(h4*)&Qs[n0 * 32 + dp] = h4{(_Float16)(q0.x * scale), (_Float16)(q0.y * scale),
                                     (_Float16)(q0.z * scale), (_Float16)(q0.w * scale)};
        *(h4*)&Qs[n1 * 32 + dp] = h4{(_Float16)(q1.x * scale), (_Float16)(q1.y * scale),
                                     (_Float16)(q1.z * scale), (_Float16)(q1.w * scale)};
        *(h4*)&Ks[n0 * 32 + d4] = h4{(_Float16)k0.x, (_Float16)k0.y, (_Float16)k0.z, (_Float16)k0.w};
        *(h4*)&Ks[n1 * 32 + d4] = h4{(_Float16)k1.x, (_Float16)k1.y, (_Float16)k1.z, (_Float16)k1.w};
        Vt[(d4 + 0) * 64 + n0] = (_Float16)v0.x;
        Vt[(d4 + 1) * 64 + n0] = (_Float16)v0.y;
        Vt[(d4 + 2) * 64 + n0] = (_Float16)v0.z;
        Vt[(d4 + 3) * 64 + n0] = (_Float16)v0.w;
        Vt[(d4 + 0) * 64 + n1] = (_Float16)v1.x;
        Vt[(d4 + 1) * 64 + n1] = (_Float16)v1.y;
        Vt[(d4 + 2) * 64 + n1] = (_Float16)v1.z;
        Vt[(d4 + 3) * 64 + n1] = (_Float16)v1.w;
    }
    __syncthreads();

    // S = Q K^T : 16 tiles, 8 waves x 2
    int wim = w_ % nWimg;
    int nwx = Hd / 7;
    int wy = wim / nwx, wx = wim % nwx;
    for (int rep = 0; rep < 2; ++rep) {
        int ti = wid * 2 + rep, tm = ti >> 2, tn = ti & 3;
        int mrow = tm * 16 + (lane & 15), half = lane >> 4;
        int ncol = tn * 16 + (lane & 15), kg = half * 16;
        v16h a  = *(const v16h*)&Qs[mrow * 32 + half * 16];
        v16h bf = *(const v16h*)&Ks[ncol * 32 + kg];
        v8f acc = {};
        acc = WMMA16(a, bf, acc);
#pragma unroll
        for (int vv = 0; vv < 8; ++vv) {
            int i = tm * 16 + vv + 8 * half;
            int j = tn * 16 + (lane & 15);
            float s = acc[vv];
            if (i < 49 && j < 49) {
                int ih = i / 7, iw = i % 7, jh = j / 7, jw = j % 7;
                int rel = (ih - jh + 6) * 13 + (iw - jw + 6);
                s += relb[rel * nh + h];
                if (shift > 0) {
                    int yi = wy * 7 + ih, xi = wx * 7 + iw;
                    int yj = wy * 7 + jh, xj = wx * 7 + jw;
                    int ri = (yi < Hd - 7 ? 0 : (yi < Hd - shift ? 1 : 2)) * 3 +
                             (xi < Hd - 7 ? 0 : (xi < Hd - shift ? 1 : 2));
                    int rj = (yj < Hd - 7 ? 0 : (yj < Hd - shift ? 1 : 2)) * 3 +
                             (xj < Hd - 7 ? 0 : (xj < Hd - shift ? 1 : 2));
                    if (ri != rj) s += -100.f;
                }
            }
            Sf[i][j] = s;
        }
    }
    __syncthreads();

    // softmax rows (fp32); write P row-major, zero-padded to 64
    if (tid < 64) {
        float mx = -1e30f;
        for (int j = 0; j < 49; ++j) mx = fmaxf(mx, Sf[tid][j]);
        float sum = 0.f;
        for (int j = 0; j < 49; ++j) { float e = __expf(Sf[tid][j] - mx); Sf[tid][j] = e; sum += e; }
        float inv = 1.f / sum;
        for (int j = 0; j < 49; ++j)
            Pf[tid * 64 + j] = (_Float16)(Sf[tid][j] * inv);
        for (int j = 49; j < 64; ++j)
            Pf[tid * 64 + j] = (_Float16)0.f;
    }
    __syncthreads();

    // O = P V : 8 tiles (4x2), one per wave, 2 k-steps
    {
        int tm = wid >> 1, tn = wid & 1;
        int mrow = tm * 16 + (lane & 15), half = lane >> 4;
        int ncol = tn * 16 + (lane & 15), kg = half * 16;
        v8f acc = {};
#pragma unroll
        for (int ks = 0; ks < 2; ++ks) {
            v16h a  = *(const v16h*)&Pf[mrow * 64 + ks * 32 + half * 16];
            v16h bf = *(const v16h*)&Vt[ncol * 64 + ks * 32 + kg];
            acc = WMMA16(a, bf, acc);
        }
#pragma unroll
        for (int vv = 0; vv < 8; ++vv) {
            int i = tm * 16 + vv + 8 * half;
            if (i < 49) out[(size_t)(w_ * 49 + i) * C + h * 32 + ncol] = acc[vv];
        }
    }
}

// ---------------------------------------------------------------- merge / expand rearranges
__global__ void merge_gather_kernel(const float* __restrict__ x, float* __restrict__ cat,
                                    int Hd, int C) {
    int H2 = Hd / 2, C4 = 4 * C;
    int n_tot = BATCH * H2 * H2 * C4;
    int idx = blockIdx.x * 256 + threadIdx.x;
    if (idx >= n_tot) return;
    int c4 = idx % C4, t = idx / C4;
    int w2 = t % H2, h2 = (t / H2) % H2, b = t / (H2 * H2);
    int k = c4 / C, c = c4 % C;
    int i = k & 1, j = k >> 1;   // order: (0,0),(1,0),(0,1),(1,1)
    cat[idx] = x[((size_t)(b * Hd + 2 * h2 + i) * Hd + 2 * w2 + j) * C + c];
}

// t: (B,H,W,2*Cin) -> out: (B,2H,2W,Cin/2)
__global__ void expand_rearrange_kernel(const float* __restrict__ t, float* __restrict__ out,
                                        int Hd, int Cin) {
    int Co = Cin / 2, H2 = Hd * 2;
    int n_tot = BATCH * H2 * H2 * Co;
    int idx = blockIdx.x * 256 + threadIdx.x;
    if (idx >= n_tot) return;
    int co = idx % Co, tt = idx / Co;
    int xo = tt % H2, yo = (tt / H2) % H2, b = tt / (H2 * H2);
    int hc = yo >> 1, p = yo & 1, wc = xo >> 1, q = xo & 1;
    out[idx] = t[((size_t)(b * Hd + hc) * Hd + wc) * (2 * Cin) + (p * 2 + q) * Co + co];
}

// ---------------------------------------------------------------- loss
__global__ void loss_kernel(const float* __restrict__ pred, const float* __restrict__ maskBL,
                            const float* __restrict__ x, float* accum) {
    int idx = blockIdx.x * 256 + threadIdx.x;
    if (idx >= BATCH * L0) return;
    int l = idx % L0, b = idx / L0;
    int hh = l / 56, ww = l % 56;
    const float* pr = pred + (size_t)idx * PREDC;
    float s = 0.f;
    for (int p = 0; p < 4; ++p)
        for (int q = 0; q < 4; ++q)
            for (int c = 0; c < 3; ++c) {
                float tg = x[((size_t)(b * 3 + c) * 224 + hh * 4 + p) * 224 + ww * 4 + q];
                float d = pr[(p * 4 + q) * 3 + c] - tg;
                s += d * d;
            }
    s *= (1.f / 48.f);
    float m = maskBL[idx];
    atomicAdd(&accum[0], s * m);
    atomicAdd(&accum[1], m);
}

__global__ void finalize_kernel(const float* accum, float* out) {
    if (threadIdx.x == 0 && blockIdx.x == 0) out[0] = accum[0] / accum[1];
}

// ---------------------------------------------------------------- host orchestration
struct Blk {
    const float *fc1_b, *fc1_w, *fc2_b, *fc2_w, *n1_b, *n1_g, *n2_b, *n2_g,
                *proj_b, *proj_w, *qkv_b, *qkv_w, *rel_bias;
};

extern "C" void kernel_launch(void* const* d_in, const int* in_sizes, int n_in,
                              void* d_out, int out_size, void* d_ws, size_t ws_size,
                              hipStream_t stream) {
    (void)in_sizes; (void)n_in; (void)out_size; (void)ws_size;
    auto F = [&](int i) { return (const float*)d_in[i]; };

    // ---- unpack inputs (x, window_arr, noise, then params in jax sorted-key tree order)
    const float* x_in  = F(0);
    const float* noise = F(2);
    int pi = 3;
    const float* fe_exp_w = F(pi++);  // first_expand: exp_w, n_b, n_g
    const float* fe_n_b   = F(pi++);
    const float* fe_n_g   = F(pi++);
    auto readBlk = [&]() {
        Blk k;
        k.fc1_b = F(pi++); k.fc1_w = F(pi++); k.fc2_b = F(pi++); k.fc2_w = F(pi++);
        k.n1_b = F(pi++);  k.n1_g = F(pi++);  k.n2_b = F(pi++);  k.n2_g = F(pi++);
        k.proj_b = F(pi++); k.proj_w = F(pi++); k.qkv_b = F(pi++); k.qkv_w = F(pi++);
        k.rel_bias = F(pi++);
        return k;
    };
    Blk enc[4][2];
    const float *mg_n_b[3], *mg_n_g[3], *mg_red[3];
    for (int s = 0; s < 4; ++s) {
        enc[s][0] = readBlk(); enc[s][1] = readBlk();
        if (s < 3) { mg_n_b[s] = F(pi++); mg_n_g[s] = F(pi++); mg_red[s] = F(pi++); }
    }
    Blk dec[3][2];
    const float *ex_w[2], *ex_n_b[2], *ex_n_g[2];
    for (int s = 0; s < 3; ++s) {
        dec[s][0] = readBlk(); dec[s][1] = readBlk();
        if (s < 2) { ex_w[s] = F(pi++); ex_n_b[s] = F(pi++); ex_n_g[s] = F(pi++); }
    }
    const float* mask_token = F(pi++);
    const float* nup_b = F(pi++); const float* nup_g = F(pi++);
    const float* patch_b = F(pi++); const float* patch_n_b = F(pi++);
    const float* patch_n_g = F(pi++); const float* patch_w = F(pi++);
    const float* pred_b = F(pi++);  const float* pred_w = F(pi++);

    // ---- workspace slabs (floats)
    float* ws = (float*)d_ws;
    float* hB   = ws;                 // activation (B,H,W,C)
    float* t0   = ws + 10000000;      // scratch M x C
    float* t1   = ws + 20000000;      // scratch M x C
    float* qkvB = ws + 30000000;      // M x 3C
    float* eB   = ws + 60000000;      // M x 4C
    float* acc  = ws + 99000000;      // 2 floats

    float* outF   = (float*)d_out;
    float* predB  = outF + 1;
    float* maskBL = outF + 1 + (size_t)BATCH * L0 * PREDC;

    auto g1 = [](int n) { return dim3((unsigned)((n + 255) / 256)); };
    auto runLN = [&](const float* in, const float* g, const float* b, float* out, int M, int C) {
        ln_kernel<<<dim3((unsigned)((M + 7) / 8)), 256, 0, stream>>>(in, g, b, out, M, C);
    };
    auto runGemm = [&](const float* A, const float* B, const float* bias, const float* resid,
                       float* C, int M, int N, int K, int act) {
        gemm_kernel<<<dim3((unsigned)((M + 63) / 64), (unsigned)((N + 63) / 64)), 256, 0, stream>>>(
            A, B, bias, resid, C, M, N, K, act);
    };
    auto runBlock = [&](const Blk& bp, int nh, int Hd, int C, int shift) {
        int M = BATCH * Hd * Hd;
        runLN(hB, bp.n1_g, bp.n1_b, t0, M, C);
        part_kernel<<<g1(M * C), 256, 0, stream>>>(t0, t1, Hd, C, shift);
        runGemm(t1, bp.qkv_w, bp.qkv_b, nullptr, qkvB, M, 3 * C, C, 0);
        int nWimg = (Hd / 7) * (Hd / 7);
        attn_kernel<<<dim3((unsigned)(BATCH * nWimg * nh)), 256, 0, stream>>>(
            qkvB, bp.rel_bias, t0, nh, nWimg, Hd, shift);
        runGemm(t0, bp.proj_w, bp.proj_b, nullptr, t1, M, C, C, 0);
        rev_add_kernel<<<g1(M * C), 256, 0, stream>>>(hB, t1, t0, Hd, C, shift);
        runLN(t0, bp.n2_g, bp.n2_b, t1, M, C);
        runGemm(t1, bp.fc1_w, bp.fc1_b, nullptr, eB, M, 4 * C, C, 1);   // fused GELU
        runGemm(eB, bp.fc2_w, bp.fc2_b, t0, hB, M, C, 4 * C, 0);        // fused residual
    };

    // ---- patch embed + LN + masking
    patch_embed_kernel<<<g1(BATCH * L0 * 96), 256, 0, stream>>>(x_in, patch_w, patch_b, t0);
    runLN(t0, patch_n_g, patch_n_b, t1, BATCH * L0, 96);
    fill_kernel<<<g1(BATCH * L0), 256, 0, stream>>>(maskBL, 1.f, BATCH * L0);
    mask_rank_kernel<<<dim3(BATCH), 256, 0, stream>>>(noise, maskBL);
    blend_kernel<<<g1(BATCH * L0 * 96), 256, 0, stream>>>(t1, mask_token, maskBL, hB);

    // ---- encoder
    static const int heads[4] = {3, 6, 12, 24};
    int Hd = 56, C = 96;
    for (int s = 0; s < 4; ++s) {
        for (int j = 0; j < 2; ++j) {
            int shift = (j == 1 && Hd > 7) ? 3 : 0;
            runBlock(enc[s][j], heads[s], Hd, C, shift);
        }
        if (s < 3) {
            int H2 = Hd / 2, M2 = BATCH * H2 * H2;
            merge_gather_kernel<<<g1(M2 * 4 * C), 256, 0, stream>>>(hB, t0, Hd, C);
            runLN(t0, mg_n_g[s], mg_n_b[s], t1, M2, 4 * C);
            runGemm(t1, mg_red[s], nullptr, nullptr, hB, M2, 2 * C, 4 * C, 0);
            Hd = H2; C *= 2;
        }
    }

    // ---- first expand: (B,7,7,768) -> (B,14,14,384)
    runGemm(hB, fe_exp_w, nullptr, nullptr, t0, BATCH * Hd * Hd, 2 * C, C, 0);
    expand_rearrange_kernel<<<g1(BATCH * Hd * Hd * 2 * C), 256, 0, stream>>>(t0, t1, Hd, C);
    runLN(t1, fe_n_g, fe_n_b, hB, BATCH * (2 * Hd) * (2 * Hd), C / 2);
    Hd *= 2; C /= 2;

    // ---- decoder
    for (int s = 0; s < 3; ++s) {
        int nh = heads[2 - s];
        for (int j = 0; j < 2; ++j) {
            int shift = (j == 1 && Hd > 7) ? 3 : 0;
            runBlock(dec[s][j], nh, Hd, C, shift);
        }
        if (s < 2) {
            int M = BATCH * Hd * Hd;
            runGemm(hB, ex_w[s], nullptr, nullptr, t0, M, 2 * C, C, 0);
            expand_rearrange_kernel<<<g1(M * 2 * C), 256, 0, stream>>>(t0, t1, Hd, C);
            runLN(t1, ex_n_g[s], ex_n_b[s], hB, BATCH * (2 * Hd) * (2 * Hd), C / 2);
            Hd *= 2; C /= 2;
        }
    }

    // ---- head + loss
    runLN(hB, nup_g, nup_b, t0, BATCH * L0, 96);
    runGemm(t0, pred_w, pred_b, nullptr, predB, BATCH * L0, PREDC, 96, 0);
    fill_kernel<<<dim3(1), 256, 0, stream>>>(acc, 0.f, 2);
    loss_kernel<<<g1(BATCH * L0), 256, 0, stream>>>(predB, maskBL, x_in, acc);
    finalize_kernel<<<dim3(1), 32, 0, stream>>>(acc, outF);
}
